// CTRNN_67250597920776
// MI455X (gfx1250) — matrix operand order repaired and checked by reading, not compiled, not run on hardware
//
#include <hip/hip_runtime.h>

typedef __attribute__((ext_vector_type(16))) __bf16 v16bf;
typedef __attribute__((ext_vector_type(8)))  float  v8f;
typedef __attribute__((ext_vector_type(4)))  unsigned int v4u;

#define DEVINL __device__ __forceinline__

// ---- bf16 helpers --------------------------------------------------------
// pair pack: prefer HW v_cvt_pk_bf16_f32; else round-half-up + v_perm_b32
#if __has_builtin(__builtin_amdgcn_cvt_pk_bf16_f32)
DEVINL unsigned bfpack(float lo, float hi) {
  typedef __attribute__((ext_vector_type(2))) __bf16 v2bf;
  union { v2bf v; unsigned u; } r;
  r.v = __builtin_amdgcn_cvt_pk_bf16_f32(lo, hi);
  return r.u;
}
#else
DEVINL unsigned bfpack(float lo, float hi) {
  unsigned a = __float_as_uint(lo) + 0x8000u;
  unsigned b = __float_as_uint(hi) + 0x8000u;
  return __builtin_amdgcn_perm(b, a, 0x07060302u);  // {b.hi16, a.hi16}
}
#endif
// scalar: round-half-up; (x>>16)+trunc store selects ds_store_b16_d16_hi
DEVINL void store_bf16(unsigned short* p, float x) {
  *p = (unsigned short)((__float_as_uint(x) + 0x8000u) >> 16);
}

// ---- activation ----------------------------------------------------------
#if __has_builtin(__builtin_amdgcn_tanhf)
DEVINL float act(float z) { return __builtin_amdgcn_tanhf(z); }
#elif __has_builtin(__builtin_amdgcn_tanh_f32)
DEVINL float act(float z) { return __builtin_amdgcn_tanh_f32(z); }
#else
// saturation-safe: 1 - 2/(e^{2z}+1); inf-safe at both ends
DEVINL float act(float z) {
  float e = __expf(2.0f * z);
  return 1.0f - 2.0f * __builtin_amdgcn_rcpf(e + 1.0f);
}
#endif

union BF16x16 { v4u q[2]; v16bf v; unsigned u[8]; };

DEVINL v8f wmma_bf16(v16bf a, v16bf b, v8f c) {
  // D = A(16x32 bf16) x B(32x16 bf16) + C(16x16 f32)
  return __builtin_amdgcn_wmma_f32_16x16x32_bf16(
      false, a, false, b, (short)0, c, false, false);
}

// B-matrix tile from staged LDS: per lane 8 consecutive uints (32B)
DEVINL v16bf load_B(const unsigned* wbuf, int tileIdx, int lane) {
  const unsigned* p = wbuf + tileIdx * 256 + lane * 8;
  BF16x16 r;
  r.q[0] = *reinterpret_cast<const v4u*>(p);
  r.q[1] = *reinterpret_cast<const v4u*>(p + 4);
  return r.v;
}

#define XROW 80  // padded row stride (shorts): 160B keeps 16B align, spreads banks

// A-matrix tile (16xK, K-block kb) from per-wave bf16 state buffer.
DEVINL v16bf load_A_lds(const unsigned short* xbuf, int lane, int kb) {
  int lo = lane & 15, hi = lane >> 4;
  int off = 32 * kb + 16 * hi;
  const unsigned short* p = xbuf + lo * XROW + off;
  BF16x16 r;
  r.q[0] = *reinterpret_cast<const v4u*>(p);
  r.q[1] = *reinterpret_cast<const v4u*>(p + 8);
  return r.v;
}

// One dynamics evaluation: k = tanh(xi @ Wdyn + (b_dyn + te*Wt)) + xi*(-1/tau)
DEVINL void dyn_eval(const v8f* xi, float te, v8f* k,
                     unsigned short* xbuf, int lane,
                     const v16bf wd[2][4],
                     const float* bd, const float* wt, const float* ntau) {
  const int lo = lane & 15, hi = lane >> 4;
  // dump state to LDS as bf16 (per-wave private region; LDS pipe is in-order)
  #pragma unroll
  for (int nb = 0; nb < 4; ++nb) {
    #pragma unroll
    for (int r = 0; r < 8; ++r)
      store_bf16(&xbuf[(r + 8 * hi) * XROW + 16 * nb + lo], xi[nb][r]);
  }
  v16bf a0 = load_A_lds(xbuf, lane, 0);
  v16bf a1 = load_A_lds(xbuf, lane, 1);
  #pragma unroll
  for (int nb = 0; nb < 4; ++nb) {
    float cb = fmaf(te, wt[nb], bd[nb]);   // time-column contribution + bias
    v8f z = {};                            // zero C (inline-0 / loop-invariant)
    z = wmma_bf16(a0, wd[0][nb], z);
    z = wmma_bf16(a1, wd[1][nb], z);
    #pragma unroll
    for (int r = 0; r < 8; ++r)
      k[nb][r] = act(z[r] + cb) + xi[nb][r] * ntau[nb];
  }
}

__global__ __launch_bounds__(256)
void ctrnn_rk4_wmma(const float* __restrict__ inputs,
                    const float* __restrict__ W_in,
                    const float* __restrict__ b_in,
                    const float* __restrict__ W_dyn,
                    const float* __restrict__ b_dyn,
                    const float* __restrict__ tau,
                    const float* __restrict__ W_out,
                    const float* __restrict__ b_out,
                    float* __restrict__ out, int nTiles) {
  __shared__ unsigned lds_win[16 * 256];         // 16 B-tiles of W_in  (16KB)
  __shared__ unsigned lds_wd[8 * 256];           //  8 B-tiles of W_dyn ( 8KB)
  __shared__ unsigned lds_wo[2 * 256];           //  2 B-tiles of W_out ( 2KB)
  __shared__ unsigned short lds_x[8][16 * XROW]; // per-wave state scratch

  const int tid = threadIdx.x;

  // ---- stage all weights as bf16 B-matrix tiles in LDS (once per block) --
  for (int idx = tid; idx < 16 * 256; idx += 256) {
    int j = idx & 7, lane = (idx >> 3) & 31, t = idx >> 8;
    int kb = t >> 2, nb = t & 3;
    int k = 32 * kb + ((lane & 16) ? 16 : 0) + 2 * j;
    int col = 16 * nb + (lane & 15);
    lds_win[idx] = bfpack(W_in[k * 64 + col], W_in[(k + 1) * 64 + col]);
  }
  for (int idx = tid; idx < 8 * 256; idx += 256) {
    int j = idx & 7, lane = (idx >> 3) & 31, t = idx >> 8;
    int kb = t >> 2, nb = t & 3;
    int k = 32 * kb + ((lane & 16) ? 16 : 0) + 2 * j;
    int col = 16 * nb + (lane & 15);
    lds_wd[idx] = bfpack(W_dyn[k * 64 + col], W_dyn[(k + 1) * 64 + col]);
  }
  for (int idx = tid; idx < 2 * 256; idx += 256) {
    int j = idx & 7, lane = (idx >> 3) & 31, nb = idx >> 8;
    int k = ((lane & 16) ? 16 : 0) + 2 * j;
    int col = 16 * nb + (lane & 15);
    lds_wo[idx] = bfpack(W_out[k * 32 + col], W_out[(k + 1) * 32 + col]);
  }
  __syncthreads();

  const int lane = tid & 31;
  const int wv = tid >> 5;
  const int lo = lane & 15, hi = lane >> 4;
  unsigned short* xbuf = lds_x[wv];

  // W_dyn B-tiles register-resident for the whole RK4 loop (64 VGPRs bf16)
  v16bf wd[2][4];
  #pragma unroll
  for (int kb = 0; kb < 2; ++kb)
    #pragma unroll
    for (int nb = 0; nb < 4; ++nb)
      wd[kb][nb] = load_B(lds_wd, kb * 4 + nb, lane);

  // per-lane (per hidden column) scalars
  float bd[4], wt[4], ntau[4], binv[4];
  #pragma unroll
  for (int nb = 0; nb < 4; ++nb) {
    int col = 16 * nb + lo;
    bd[nb]   = b_dyn[col];
    wt[nb]   = W_dyn[64 * 64 + col];   // time row of W_dyn ([65,64] row-major)
    ntau[nb] = -1.0f / tau[col];
    binv[nb] = b_in[col];
  }
  float bo[2];
  bo[0] = b_out[lo];
  bo[1] = b_out[16 + lo];

  const float dt = 0.1f, hdt = 0.05f;
  const float dt6 = dt / 6.0f, dt3 = dt / 3.0f;
  const int waveGlobal = blockIdx.x * 8 + wv;
  const int waveStride = gridDim.x * 8;
  float* hid = out + (size_t)nTiles * 16 * 32;

  for (int tile = waveGlobal; tile < nTiles; tile += waveStride) {
    const int row0 = tile * 16;
    const float* inrow = inputs + (size_t)(row0 + lo) * 128;

    // ---- x0 = tanh(inputs @ W_in + b_in): A from global, B from LDS ------
    v8f D[4] = {{}, {}, {}, {}};
    #pragma unroll
    for (int kb = 0; kb < 4; ++kb) {
      int off = 32 * kb + 16 * hi;
      float4 f0 = *reinterpret_cast<const float4*>(inrow + off);
      float4 f1 = *reinterpret_cast<const float4*>(inrow + off + 4);
      float4 f2 = *reinterpret_cast<const float4*>(inrow + off + 8);
      float4 f3 = *reinterpret_cast<const float4*>(inrow + off + 12);
      BF16x16 a;
      a.u[0] = bfpack(f0.x, f0.y); a.u[1] = bfpack(f0.z, f0.w);
      a.u[2] = bfpack(f1.x, f1.y); a.u[3] = bfpack(f1.z, f1.w);
      a.u[4] = bfpack(f2.x, f2.y); a.u[5] = bfpack(f2.z, f2.w);
      a.u[6] = bfpack(f3.x, f3.y); a.u[7] = bfpack(f3.z, f3.w);
      #pragma unroll
      for (int nb = 0; nb < 4; ++nb) {
        v16bf b = load_B(lds_win, kb * 4 + nb, lane);
        D[nb] = wmma_bf16(a.v, b, D[nb]);
      }
    }
    v8f x[4];
    #pragma unroll
    for (int nb = 0; nb < 4; ++nb)
      #pragma unroll
      for (int r = 0; r < 8; ++r) x[nb][r] = act(D[nb][r] + binv[nb]);

    // prefetch next tile's input panel (32 distinct 256B lines = 8KB)
    if (tile + waveStride < nTiles) {
      const float* nx = inputs + (size_t)((tile + waveStride) * 16 + lo) * 128;
      __builtin_prefetch(nx + hi * 64, 0, 1);
    }

    // ---- 10 RK4 steps, state stays in VGPRs ------------------------------
    float t0 = 0.0f;
    for (int s = 0; s < 10; ++s) {
      v8f k[4], xt[4], acc[4];
      dyn_eval(x, t0, k, xbuf, lane, wd, bd, wt, ntau);          // k1
      #pragma unroll
      for (int nb = 0; nb < 4; ++nb)
        #pragma unroll
        for (int r = 0; r < 8; ++r) {
          acc[nb][r] = fmaf(dt6, k[nb][r], x[nb][r]);
          xt[nb][r]  = fmaf(hdt, k[nb][r], x[nb][r]);
        }
      dyn_eval(xt, t0 + hdt, k, xbuf, lane, wd, bd, wt, ntau);   // k2
      #pragma unroll
      for (int nb = 0; nb < 4; ++nb)
        #pragma unroll
        for (int r = 0; r < 8; ++r) {
          acc[nb][r] = fmaf(dt3, k[nb][r], acc[nb][r]);
          xt[nb][r]  = fmaf(hdt, k[nb][r], x[nb][r]);
        }
      dyn_eval(xt, t0 + hdt, k, xbuf, lane, wd, bd, wt, ntau);   // k3
      #pragma unroll
      for (int nb = 0; nb < 4; ++nb)
        #pragma unroll
        for (int r = 0; r < 8; ++r) {
          acc[nb][r] = fmaf(dt3, k[nb][r], acc[nb][r]);
          xt[nb][r]  = fmaf(dt,  k[nb][r], x[nb][r]);
        }
      dyn_eval(xt, t0 + dt, k, xbuf, lane, wd, bd, wt, ntau);    // k4
      #pragma unroll
      for (int nb = 0; nb < 4; ++nb)
        #pragma unroll
        for (int r = 0; r < 8; ++r)
          x[nb][r] = fmaf(dt6, k[nb][r], acc[nb][r]);
      t0 += dt;
    }

    // ---- hidden = x[:, :32]; out = hidden @ W_out + b_out ----------------
    #pragma unroll
    for (int nb = 0; nb < 2; ++nb)
      #pragma unroll
      for (int r = 0; r < 8; ++r)
        store_bf16(&xbuf[(r + 8 * hi) * XROW + 16 * nb + lo], x[nb][r]);
    v16bf ah = load_A_lds(xbuf, lane, 0);   // K = 0..31 (first half of hidden)
    #pragma unroll
    for (int nb = 0; nb < 2; ++nb) {
      v8f c = {};
      v16bf b = load_B(lds_wo, nb, lane);
      c = wmma_bf16(ah, b, c);
      #pragma unroll
      for (int r = 0; r < 8; ++r)
        out[(size_t)(row0 + r + 8 * hi) * 32 + 16 * nb + lo] = c[r] + bo[nb];
    }
    #pragma unroll
    for (int nb = 0; nb < 2; ++nb)
      #pragma unroll
      for (int r = 0; r < 8; ++r)
        hid[(size_t)(row0 + r + 8 * hi) * 32 + 16 * nb + lo] = x[nb][r];
  }
}

extern "C" void kernel_launch(void* const* d_in, const int* in_sizes, int n_in,
                              void* d_out, int out_size, void* d_ws, size_t ws_size,
                              hipStream_t stream) {
  (void)n_in; (void)out_size; (void)d_ws; (void)ws_size;
  const float* inputs = (const float*)d_in[0];
  const float* W_in   = (const float*)d_in[1];
  const float* b_in   = (const float*)d_in[2];
  const float* W_dyn  = (const float*)d_in[3];
  const float* b_dyn  = (const float*)d_in[4];
  const float* tau    = (const float*)d_in[5];
  const float* W_out  = (const float*)d_in[6];
  const float* b_out  = (const float*)d_in[7];
  float* out = (float*)d_out;

  int B = in_sizes[0] / 128;
  int nTiles = B / 16;               // B = 262144 -> 16384 tiles
  int blocks = 512;                  // 8 waves/block -> 4 tiles per wave
  if (blocks * 8 > nTiles) blocks = (nTiles + 7) / 8;
  ctrnn_rk4_wmma<<<blocks, 256, 0, stream>>>(inputs, W_in, b_in, W_dyn, b_dyn,
                                             tau, W_out, b_out, out, nTiles);
}
